// CDiTBlock_7825430413788
// MI455X (gfx1250) — compile-verified, matrix operand order checked by reading, not run on hardware
//
#include <hip/hip_runtime.h>

#define B_   4
#define T_   1024
#define D_   512
#define H_   8
#define DH_  64
#define MLP_ 2048
#define M_   (B_ * T_)
#define EPS_ 1e-6f
#define KC_  128    // K-chunk staged in LDS per TDM transfer

typedef unsigned short u16;
typedef unsigned int   u32;
typedef __attribute__((ext_vector_type(16))) __bf16 v16bf;
typedef __attribute__((ext_vector_type(8)))  float  v8f;
typedef __attribute__((ext_vector_type(4)))  unsigned int u32x4;
typedef __attribute__((ext_vector_type(4)))  unsigned int v4u;
typedef __attribute__((ext_vector_type(8)))  int v8i;
typedef __attribute__((ext_vector_type(4)))  int v4i;

#if defined(__has_builtin)
#if __has_builtin(__builtin_amdgcn_tensor_load_to_lds) && __has_builtin(__builtin_amdgcn_s_wait_tensorcnt)
#define USE_TDM 1
#endif
#endif
#ifndef USE_TDM
#define USE_TDM 0
#endif

union FragU {
    v16bf v;
    u32x4 q[2];
    u32   u[8];
    u16   s[16];
};

__device__ __forceinline__ u16 f2bf(float x) {
    u32 u = __float_as_uint(x);
    u32 r = u + 0x7FFFu + ((u >> 16) & 1u);   // round-to-nearest-even
    return (u16)(r >> 16);
}

__device__ __forceinline__ v16bf load_frag(const u16* p) {
    FragU f;
    f.q[0] = *reinterpret_cast<const u32x4*>(p);        // elements 0..7  (K: kh .. kh+7)
    f.q[1] = *reinterpret_cast<const u32x4*>(p + 16);   // elements 8..15 (K: kh+16 .. kh+23)
    return f.v;
}

__device__ __forceinline__ v16bf neg_frag(v16bf a) {
    FragU f; f.v = a;
#pragma unroll
    for (int i = 0; i < 8; ++i) f.u[i] ^= 0x80008000u;
    return f.v;
}

__device__ __forceinline__ v8f wmma_bf16(v16bf a, v16bf b, v8f c) {
    return __builtin_amdgcn_wmma_f32_16x16x32_bf16(false, a, false, b, (short)0, c, false, false);
}

__device__ __forceinline__ float gelu_tanh(float x) {
    const float c = 0.7978845608028654f; // sqrt(2/pi)
    float x3 = x * x * x;
    return 0.5f * x * (1.0f + tanhf(c * (x + 0.044715f * x3)));
}

#if USE_TDM
// ---------------------------------------------------------------------------
// TDM: DMA a (64 rows x KC_ cols) bf16 tile of a row-major (Nrows x K) matrix
// into LDS. 2D descriptor: tensor_dim0 = K (contiguous), dim0 stride = K,
// tile_dim0 = KC_, tile_dim1 = 64, data_size = 2 bytes, type = 2 (image).
// ---------------------------------------------------------------------------
__device__ __forceinline__ void tdm_issue(const u16* ldsdst, const u16* gsrc,
                                          int K, int Nrows) {
    unsigned lds_addr = (unsigned)(uintptr_t)ldsdst;
    unsigned long long ga = (unsigned long long)(uintptr_t)gsrc;
    v4u g0;
    g0[0] = 1u;                                            // count=1, user D#
    g0[1] = lds_addr;                                      // LDS byte address
    g0[2] = (unsigned)ga;                                  // global_addr[31:0]
    g0[3] = (unsigned)((ga >> 32) & 0x01FFFFFFull) | 0x80000000u; // [56:32] | type=2
    v8i g1;
    g1[0] = 0x00010000;                                    // data_size=1 (2 bytes)
    g1[1] = (int)(((unsigned)K & 0xFFFFu) << 16);          // tensor_dim0 lo16
    g1[2] = (int)((((unsigned)K >> 16) & 0xFFFFu) |
                  (((unsigned)Nrows & 0xFFFFu) << 16));    // dim0 hi16 | dim1 lo16
    g1[3] = (int)((((unsigned)Nrows >> 16) & 0xFFFFu) |
                  ((unsigned)KC_ << 16));                  // dim1 hi16 | tile_dim0
    g1[4] = 64;                                            // tile_dim1=64, tile_dim2=0
    g1[5] = K;                                             // tensor_dim0_stride lo32
    g1[6] = 0;                                             // stride hi16, dim1_stride lo
    g1[7] = 0;
    v4i gz = {0, 0, 0, 0};
#if __clang_major__ >= 23
    v8i gz8 = {0, 0, 0, 0, 0, 0, 0, 0};
    __builtin_amdgcn_tensor_load_to_lds(g0, g1, gz, gz, gz8, 0);
#else
    __builtin_amdgcn_tensor_load_to_lds(g0, g1, gz, gz, 0);
#endif
}
#endif

// ---------------------------------------------------------------------------
// fp32 -> bf16 conversion
// ---------------------------------------------------------------------------
__global__ void conv_bf16_kernel(const float* __restrict__ in, u16* __restrict__ out, int n) {
    int i = blockIdx.x * 256 + threadIdx.x;
    if (i < n) out[i] = f2bf(in[i]);
}

// ---------------------------------------------------------------------------
// silu of conditioning vector
// ---------------------------------------------------------------------------
__global__ void silu_kernel(const float* __restrict__ cr, const float* __restrict__ ci,
                            float* __restrict__ sr, float* __restrict__ si) {
    int i = blockIdx.x * 256 + threadIdx.x;
    if (i < B_ * D_) {
        float a = cr[i]; sr[i] = a / (1.f + __expf(-a));
        float b = ci[i]; si[i] = b / (1.f + __expf(-b));
    }
}

// ---------------------------------------------------------------------------
// adaLN modulation: m = clinear(silu(c)); tiny (B=4 rows), plain fp32
// ---------------------------------------------------------------------------
__global__ __launch_bounds__(256) void ada_kernel(
    const float* __restrict__ sr, const float* __restrict__ si,
    const float* __restrict__ Wr, const float* __restrict__ Wi,
    const float* __restrict__ Br, const float* __restrict__ Bi,
    float* __restrict__ mr, float* __restrict__ mi)
{
    int j = blockIdx.x * 256 + threadIdx.x;   // 0 .. 6*D-1
    int b = blockIdx.y;
    const float* xr = sr + b * D_;
    const float* xi = si + b * D_;
    const float* wr = Wr + (size_t)j * D_;
    const float* wi = Wi + (size_t)j * D_;
    float ar = 0.f, ai = 0.f;
    for (int k = 0; k < D_; ++k) {
        float xrk = xr[k], xik = xi[k], wrk = wr[k], wik = wi[k];
        ar += xrk * wrk - xik * wik;
        ai += xik * wrk + xrk * wik;
    }
    mr[b * 6 * D_ + j] = ar + Br[j] - Bi[j];
    mi[b * 6 * D_ + j] = ai + Br[j] + Bi[j];
}

// ---------------------------------------------------------------------------
// complex LayerNorm + modulate -> bf16 activations
// ---------------------------------------------------------------------------
__global__ __launch_bounds__(512) void cln_mod_kernel(
    const float* __restrict__ xr, const float* __restrict__ xi,
    const float* __restrict__ mr, const float* __restrict__ mi,
    int sh_off, int sc_off,
    u16* __restrict__ hr, u16* __restrict__ hi)
{
    const int row = blockIdx.x;
    const int d   = threadIdx.x;
    const int b   = row / T_;
    const float vr = xr[(size_t)row * D_ + d];
    const float vi = xi[(size_t)row * D_ + d];

    __shared__ float red[16];
    float vals[4] = { vr, vr * vr, vi, vi * vi };
    float sums[4];
#pragma unroll
    for (int q = 0; q < 4; ++q) {
        float v = vals[q];
        v += __shfl_xor(v, 1);  v += __shfl_xor(v, 2);
        v += __shfl_xor(v, 4);  v += __shfl_xor(v, 8);  v += __shfl_xor(v, 16);
        if ((threadIdx.x & 31) == 0) red[threadIdx.x >> 5] = v;
        __syncthreads();
        if (threadIdx.x < 16) {
            float t = red[threadIdx.x];
            t += __shfl_xor(t, 1); t += __shfl_xor(t, 2);
            t += __shfl_xor(t, 4); t += __shfl_xor(t, 8);
            red[threadIdx.x] = t;
        }
        __syncthreads();
        sums[q] = red[0];
        __syncthreads();
    }
    const float inv = 1.0f / (float)D_;
    float meanR = sums[0] * inv, varR = sums[1] * inv - meanR * meanR;
    float meanI = sums[2] * inv, varI = sums[3] * inv - meanI * meanI;
    float nr = (vr - meanR) * rsqrtf(varR + EPS_);
    float ni = (vi - meanI) * rsqrtf(varI + EPS_);

    const float scr = mr[b * 6 * D_ + sc_off + d];
    const float sci = mi[b * 6 * D_ + sc_off + d];
    const float shr = mr[b * 6 * D_ + sh_off + d];
    const float shi = mi[b * 6 * D_ + sh_off + d];
    const float a = 1.0f + scr, bb = sci;
    float hrv = nr * a - ni * bb + shr;
    float hiv = nr * bb + ni * a + shi;
    hr[(size_t)row * D_ + d] = f2bf(hrv);
    hi[(size_t)row * D_ + d] = f2bf(hiv);
}

// ---------------------------------------------------------------------------
// complex GEMM:  Y = X @ W^T + bias  (complex), bf16 WMMA, f32 accumulate.
// Weight panel (64 cols x KC_ K) staged in LDS via TDM, double buffered.
// EPI: 0 plain bf16 row-major, 1 heads (b,h,t,dh) bf16, 2 heads-transposed
//      (b,h,dh,t) bf16, 3 gelu+bf16, 4 gate+residual fp32.
// ---------------------------------------------------------------------------
template<int EPI>
__global__ __launch_bounds__(256) void cgemm_kernel(
    const u16* __restrict__ Xr, const u16* __restrict__ Xi,
    const u16* __restrict__ Wr, const u16* __restrict__ Wi,
    const float* __restrict__ Br, const float* __restrict__ Bi,
    int N, int K,
    u16* __restrict__ Obr, u16* __restrict__ Obi,
    float* __restrict__ Ofr, float* __restrict__ Ofi,
    const float* __restrict__ Gr, const float* __restrict__ Gi,
    const float* __restrict__ Rr, const float* __restrict__ Ri)
{
    const int wave = threadIdx.x >> 5;
    const int lane = threadIdx.x & 31;
    const int li   = lane & 15;
    const int kh   = (lane < 16) ? 0 : 8;
    const int m0   = blockIdx.x * 32 + (wave & 1) * 16;
    const int n0   = blockIdx.y * 64 + (wave >> 1) * 16;
    const int ml   = m0 + li;
    const int nl   = n0 + li;
    const int nbase = blockIdx.y * 64;
    const int nloc  = (wave >> 1) * 16 + li;       // column within LDS panel

    __shared__ __align__(16) u16 ldsW[2][2][64][KC_];   // [buf][re/im][n][k] = 64KB

    v8f accR = {};
    v8f accI = {};
    const u16* xr = Xr + (size_t)ml * K + kh;
    const u16* xi = Xi + (size_t)ml * K + kh;

    const int nchunk = K / KC_;

#if USE_TDM
    if (wave == 0) {
        tdm_issue(&ldsW[0][0][0][0], Wr + (size_t)nbase * K, K, N);
        tdm_issue(&ldsW[0][1][0][0], Wi + (size_t)nbase * K, K, N);
    }
#endif

    for (int kc = 0; kc < nchunk; ++kc) {
        const int buf = kc & 1;
#if USE_TDM
        if (wave == 0) {
            if (kc + 1 < nchunk) {
                const int nb = buf ^ 1;
                tdm_issue(&ldsW[nb][0][0][0],
                          Wr + (size_t)nbase * K + (size_t)(kc + 1) * KC_, K, N);
                tdm_issue(&ldsW[nb][1][0][0],
                          Wi + (size_t)nbase * K + (size_t)(kc + 1) * KC_, K, N);
                __builtin_amdgcn_s_wait_tensorcnt(2);   // oldest chunk complete
            } else {
                __builtin_amdgcn_s_wait_tensorcnt(0);
            }
        }
        __syncthreads();
#else
        // cooperative fallback: stage the weight chunk with vector loads
        for (int s = threadIdx.x; s < 2048; s += 256) {
            int comp = s >> 10, rem = s & 1023;
            int n = rem >> 4, ks = (rem & 15) * 8;
            const u16* src = (comp ? Wi : Wr) +
                             (size_t)(nbase + n) * K + (size_t)kc * KC_ + ks;
            *reinterpret_cast<u32x4*>(&ldsW[buf][comp][n][ks]) =
                *reinterpret_cast<const u32x4*>(src);
        }
        __syncthreads();
#endif
        const u16* lwr = &ldsW[buf][0][nloc][kh];
        const u16* lwi = &ldsW[buf][1][nloc][kh];
        const u16* axr = xr + (size_t)kc * KC_;
        const u16* axi = xi + (size_t)kc * KC_;
#pragma unroll
        for (int kk = 0; kk < KC_; kk += 32) {
            v16bf ar  = load_frag(axr + kk);
            v16bf ai  = load_frag(axi + kk);
            v16bf br  = load_frag(lwr + kk);
            v16bf bi  = load_frag(lwi + kk);
            v16bf nbi = neg_frag(bi);
            accR = wmma_bf16(ar, br,  accR);   // + Xr Wr^T
            accR = wmma_bf16(ai, nbi, accR);   // - Xi Wi^T
            accI = wmma_bf16(ai, br,  accI);   // + Xi Wr^T
            accI = wmma_bf16(ar, bi,  accI);   // + Xr Wi^T
        }
        __syncthreads();   // panel consumed before it is overwritten
    }

    const float vbr = Br[nl];
    const float vbi = Bi[nl];
    const float biasR = vbr - vbi;
    const float biasI = vbr + vbi;
    const int roff = kh;
    const int b    = m0 / T_;

    if (EPI == 2) {
        // V: store transposed heads layout (b,h,dh,t); 8 consecutive t per lane.
        const int h = nl >> 6, dh = nl & 63;
        const int t0 = m0 - b * T_ + roff;
        FragU pr, pi;
#pragma unroll
        for (int r = 0; r < 8; ++r) {
            pr.s[r] = f2bf(accR[r] + biasR);
            pi.s[r] = f2bf(accI[r] + biasI);
        }
        size_t base = ((size_t)(b * H_ + h) * DH_ + dh) * T_ + t0;
        *reinterpret_cast<u32x4*>(Obr + base) = pr.q[0];
        *reinterpret_cast<u32x4*>(Obi + base) = pi.q[0];
    } else {
#pragma unroll
        for (int r = 0; r < 8; ++r) {
            const int row = m0 + roff + r;
            float yr = accR[r] + biasR;
            float yi = accI[r] + biasI;
            if (EPI == 0) {
                size_t idx = (size_t)row * N + nl;
                Obr[idx] = f2bf(yr);
                Obi[idx] = f2bf(yi);
            } else if (EPI == 1) {
                const int h = nl >> 6, dh = nl & 63;
                const int t = row - b * T_;
                size_t idx = ((size_t)(b * H_ + h) * T_ + t) * DH_ + dh;
                Obr[idx] = f2bf(yr);
                Obi[idx] = f2bf(yi);
            } else if (EPI == 3) {
                size_t idx = (size_t)row * N + nl;
                Obr[idx] = f2bf(gelu_tanh(yr));
                Obi[idx] = f2bf(gelu_tanh(yi));
            } else if (EPI == 4) {
                size_t idx = (size_t)row * N + nl;
                float ga = Gr[b * 6 * D_ + nl];
                float gb = Gi[b * 6 * D_ + nl];
                Ofr[idx] = Rr[idx] + yr * ga - yi * gb;
                Ofi[idx] = Ri[idx] + yr * gb + yi * ga;
            }
        }
    }
}

// ---------------------------------------------------------------------------
// complex flash attention: two softmaxes (re/im scores), combine with V.
// Q,K in (b,h,t,dh) bf16; V in (b,h,dh,t) bf16. Output (b,t,D) bf16.
// ---------------------------------------------------------------------------
__global__ __launch_bounds__(256) void cattn_kernel(
    const u16* __restrict__ Qr, const u16* __restrict__ Qi,
    const u16* __restrict__ Kr, const u16* __restrict__ Ki,
    const u16* __restrict__ Vr, const u16* __restrict__ Vi,
    u16* __restrict__ Or, u16* __restrict__ Oi)
{
    const int bh   = blockIdx.y;
    const int b    = bh >> 3, h = bh & 7;
    const int wave = threadIdx.x >> 5;
    const int lane = threadIdx.x & 31;
    const int li   = lane & 15;
    const int kh   = (lane < 16) ? 0 : 8;
    const int roff = kh;
    const int q0   = blockIdx.x * 128 + wave * 16;

    const u16* qr  = Qr + (size_t)bh * T_ * DH_;
    const u16* qi  = Qi + (size_t)bh * T_ * DH_;
    const u16* kr  = Kr + (size_t)bh * T_ * DH_;
    const u16* ki  = Ki + (size_t)bh * T_ * DH_;
    const u16* vtr = Vr + (size_t)bh * DH_ * T_;
    const u16* vti = Vi + (size_t)bh * DH_ * T_;

    v16bf aQr[2], aQi[2];
#pragma unroll
    for (int s = 0; s < 2; ++s) {
        aQr[s] = load_frag(qr + (size_t)(q0 + li) * DH_ + s * 32 + kh);
        aQi[s] = load_frag(qi + (size_t)(q0 + li) * DH_ + s * 32 + kh);
    }

    float mR[8], lR[8], mI[8], lI[8];
#pragma unroll
    for (int r = 0; r < 8; ++r) { mR[r] = -3.0e38f; mI[r] = -3.0e38f; lR[r] = 0.f; lI[r] = 0.f; }

    v8f aRR[4] = {}, aII[4] = {}, aRI[4] = {}, aIR[4] = {};

    __shared__ __align__(16) u16 probs[8][2][16][32];   // [wave][re/im][row][key]

    for (int jt = 0; jt < T_ / 32; ++jt) {
        const int j0 = jt * 32;
        v8f sre[2], sim[2];
#pragma unroll
        for (int nt = 0; nt < 2; ++nt) {
            v8f sR = {};
            v8f sI = {};
            const u16* krow  = kr + (size_t)(j0 + nt * 16 + li) * DH_ + kh;
            const u16* kirow = ki + (size_t)(j0 + nt * 16 + li) * DH_ + kh;
#pragma unroll
            for (int s = 0; s < 2; ++s) {
                v16bf bKr = load_frag(krow  + s * 32);
                v16bf bKi = load_frag(kirow + s * 32);
                v16bf nKi = neg_frag(bKi);
                sR = wmma_bf16(aQr[s], bKr, sR);   // + Qr Kr^T
                sR = wmma_bf16(aQi[s], nKi, sR);   // - Qi Ki^T
                sI = wmma_bf16(aQr[s], bKi, sI);   // + Qr Ki^T
                sI = wmma_bf16(aQi[s], bKr, sI);   // + Qi Kr^T
            }
            sre[nt] = sR; sim[nt] = sI;
        }

        // online softmax (RE scores)
#pragma unroll
        for (int r = 0; r < 8; ++r) {
            float v0 = sre[0][r] * 0.125f;
            float v1 = sre[1][r] * 0.125f;
            float mx = fmaxf(v0, v1);
            mx = fmaxf(mx, __shfl_xor(mx, 1)); mx = fmaxf(mx, __shfl_xor(mx, 2));
            mx = fmaxf(mx, __shfl_xor(mx, 4)); mx = fmaxf(mx, __shfl_xor(mx, 8));
            float mn = fmaxf(mR[r], mx);
            float al = __expf(mR[r] - mn);
            float p0 = __expf(v0 - mn);
            float p1 = __expf(v1 - mn);
            float ps = p0 + p1;
            ps += __shfl_xor(ps, 1); ps += __shfl_xor(ps, 2);
            ps += __shfl_xor(ps, 4); ps += __shfl_xor(ps, 8);
            lR[r] = lR[r] * al + ps;
            mR[r] = mn;
#pragma unroll
            for (int n = 0; n < 4; ++n) { aRR[n][r] *= al; aRI[n][r] *= al; }
            probs[wave][0][roff + r][li]      = f2bf(p0);
            probs[wave][0][roff + r][16 + li] = f2bf(p1);
        }
        // online softmax (IM scores)
#pragma unroll
        for (int r = 0; r < 8; ++r) {
            float v0 = sim[0][r] * 0.125f;
            float v1 = sim[1][r] * 0.125f;
            float mx = fmaxf(v0, v1);
            mx = fmaxf(mx, __shfl_xor(mx, 1)); mx = fmaxf(mx, __shfl_xor(mx, 2));
            mx = fmaxf(mx, __shfl_xor(mx, 4)); mx = fmaxf(mx, __shfl_xor(mx, 8));
            float mn = fmaxf(mI[r], mx);
            float al = __expf(mI[r] - mn);
            float p0 = __expf(v0 - mn);
            float p1 = __expf(v1 - mn);
            float ps = p0 + p1;
            ps += __shfl_xor(ps, 1); ps += __shfl_xor(ps, 2);
            ps += __shfl_xor(ps, 4); ps += __shfl_xor(ps, 8);
            lI[r] = lI[r] * al + ps;
            mI[r] = mn;
#pragma unroll
            for (int n = 0; n < 4; ++n) { aII[n][r] *= al; aIR[n][r] *= al; }
            probs[wave][1][roff + r][li]      = f2bf(p0);
            probs[wave][1][roff + r][16 + li] = f2bf(p1);
        }
        __syncthreads();

        v16bf aPr = load_frag(&probs[wave][0][li][kh]);
        v16bf aPi = load_frag(&probs[wave][1][li][kh]);

#pragma unroll
        for (int nv = 0; nv < 4; ++nv) {
            v16bf bVr = load_frag(vtr + (size_t)(nv * 16 + li) * T_ + j0 + kh);
            v16bf bVi = load_frag(vti + (size_t)(nv * 16 + li) * T_ + j0 + kh);
            aRR[nv] = wmma_bf16(aPr, bVr, aRR[nv]);
            aII[nv] = wmma_bf16(aPi, bVi, aII[nv]);
            aRI[nv] = wmma_bf16(aPr, bVi, aRI[nv]);
            aIR[nv] = wmma_bf16(aPi, bVr, aIR[nv]);
        }
        __syncthreads();
    }

#pragma unroll
    for (int nv = 0; nv < 4; ++nv) {
        const int dh = nv * 16 + li;
#pragma unroll
        for (int r = 0; r < 8; ++r) {
            const int t = q0 + roff + r;
            float invR = 1.0f / lR[r];
            float invI = 1.0f / lI[r];
            float orv = aRR[nv][r] * invR - aII[nv][r] * invI;
            float oiv = aRI[nv][r] * invR + aIR[nv][r] * invI;
            size_t idx = (size_t)(b * T_ + t) * D_ + h * DH_ + dh;
            Or[idx] = f2bf(orv);
            Oi[idx] = f2bf(oiv);
        }
    }
}

// ---------------------------------------------------------------------------
// host launcher
// ---------------------------------------------------------------------------
extern "C" void kernel_launch(void* const* d_in, const int* in_sizes, int n_in,
                              void* d_out, int out_size, void* d_ws, size_t ws_size,
                              hipStream_t stream)
{
    (void)in_sizes; (void)n_in; (void)out_size; (void)ws_size;

    const float* x_re = (const float*)d_in[0];
    const float* x_im = (const float*)d_in[1];
    const float* c_re = (const float*)d_in[2];
    const float* c_im = (const float*)d_in[3];

    char* ws = (char*)d_ws;
    size_t off = 0;
    auto alloc = [&](size_t bytes) -> void* {
        void* p = ws + off;
        off = (off + bytes + 255) & ~(size_t)255;
        return p;
    };

    const size_t DD  = (size_t)D_ * D_;
    const size_t DM  = (size_t)D_ * MLP_;
    const size_t MD  = (size_t)M_ * D_;
    const size_t MM  = (size_t)M_ * MLP_;

    u16* wq_r  = (u16*)alloc(DD * 2);  u16* wq_i  = (u16*)alloc(DD * 2);
    u16* wk_r  = (u16*)alloc(DD * 2);  u16* wk_i  = (u16*)alloc(DD * 2);
    u16* wv_r  = (u16*)alloc(DD * 2);  u16* wv_i  = (u16*)alloc(DD * 2);
    u16* wo_r  = (u16*)alloc(DD * 2);  u16* wo_i  = (u16*)alloc(DD * 2);
    u16* wf1_r = (u16*)alloc(DM * 2);  u16* wf1_i = (u16*)alloc(DM * 2);
    u16* wf2_r = (u16*)alloc(DM * 2);  u16* wf2_i = (u16*)alloc(DM * 2);

    float* sil_r = (float*)alloc((size_t)B_ * D_ * 4);
    float* sil_i = (float*)alloc((size_t)B_ * D_ * 4);
    float* m_r   = (float*)alloc((size_t)B_ * 6 * D_ * 4);
    float* m_i   = (float*)alloc((size_t)B_ * 6 * D_ * 4);

    u16* h1_r = (u16*)alloc(MD * 2);  u16* h1_i = (u16*)alloc(MD * 2);
    u16* qh_r = (u16*)alloc(MD * 2);  u16* qh_i = (u16*)alloc(MD * 2);
    u16* kh_r = (u16*)alloc(MD * 2);  u16* kh_i = (u16*)alloc(MD * 2);
    u16* vT_r = (u16*)alloc(MD * 2);  u16* vT_i = (u16*)alloc(MD * 2);
    u16* at_r = (u16*)alloc(MD * 2);  u16* at_i = (u16*)alloc(MD * 2);
    float* x2_r = (float*)alloc(MD * 4);
    float* x2_i = (float*)alloc(MD * 4);
    u16* h2_r = (u16*)alloc(MD * 2);  u16* h2_i = (u16*)alloc(MD * 2);
    u16* g1_r = (u16*)alloc(MM * 2);  u16* g1_i = (u16*)alloc(MM * 2);

    auto conv = [&](const void* src, u16* dst, size_t n) {
        conv_bf16_kernel<<<dim3((unsigned)((n + 255) / 256)), dim3(256), 0, stream>>>(
            (const float*)src, dst, (int)n);
    };
    conv(d_in[8],  wq_r,  DD); conv(d_in[9],  wq_i,  DD);
    conv(d_in[12], wk_r,  DD); conv(d_in[13], wk_i,  DD);
    conv(d_in[16], wv_r,  DD); conv(d_in[17], wv_i,  DD);
    conv(d_in[20], wo_r,  DD); conv(d_in[21], wo_i,  DD);
    conv(d_in[24], wf1_r, DM); conv(d_in[25], wf1_i, DM);
    conv(d_in[28], wf2_r, DM); conv(d_in[29], wf2_i, DM);

    silu_kernel<<<dim3((B_ * D_ + 255) / 256), dim3(256), 0, stream>>>(c_re, c_im, sil_r, sil_i);

    ada_kernel<<<dim3(6 * D_ / 256, B_), dim3(256), 0, stream>>>(
        sil_r, sil_i,
        (const float*)d_in[4], (const float*)d_in[5],
        (const float*)d_in[6], (const float*)d_in[7],
        m_r, m_i);

    // LN1 + modulate (sh_a = chunk 0, sc_a = chunk 1)
    cln_mod_kernel<<<dim3(M_), dim3(512), 0, stream>>>(
        x_re, x_im, m_r, m_i, 0 * D_, 1 * D_, h1_r, h1_i);

    dim3 gD(M_ / 32, D_ / 64), gM(M_ / 32, MLP_ / 64), blk(256);

    // Q, K -> heads layout; V -> transposed heads layout
    cgemm_kernel<1><<<gD, blk, 0, stream>>>(h1_r, h1_i, wq_r, wq_i,
        (const float*)d_in[10], (const float*)d_in[11], D_, D_,
        qh_r, qh_i, nullptr, nullptr, nullptr, nullptr, nullptr, nullptr);
    cgemm_kernel<1><<<gD, blk, 0, stream>>>(h1_r, h1_i, wk_r, wk_i,
        (const float*)d_in[14], (const float*)d_in[15], D_, D_,
        kh_r, kh_i, nullptr, nullptr, nullptr, nullptr, nullptr, nullptr);
    cgemm_kernel<2><<<gD, blk, 0, stream>>>(h1_r, h1_i, wv_r, wv_i,
        (const float*)d_in[18], (const float*)d_in[19], D_, D_,
        vT_r, vT_i, nullptr, nullptr, nullptr, nullptr, nullptr, nullptr);

    cattn_kernel<<<dim3(T_ / 128, B_ * H_), blk, 0, stream>>>(
        qh_r, qh_i, kh_r, kh_i, vT_r, vT_i, at_r, at_i);

    // o-projection + gate(g_a = chunk 2) + residual(x) -> x2 (fp32)
    cgemm_kernel<4><<<gD, blk, 0, stream>>>(at_r, at_i, wo_r, wo_i,
        (const float*)d_in[22], (const float*)d_in[23], D_, D_,
        nullptr, nullptr, x2_r, x2_i,
        m_r + 2 * D_, m_i + 2 * D_, x_re, x_im);

    // LN2 + modulate (sh_m = chunk 3, sc_m = chunk 4)
    cln_mod_kernel<<<dim3(M_), dim3(512), 0, stream>>>(
        x2_r, x2_i, m_r, m_i, 3 * D_, 4 * D_, h2_r, h2_i);

    // f1 + gelu
    cgemm_kernel<3><<<gM, blk, 0, stream>>>(h2_r, h2_i, wf1_r, wf1_i,
        (const float*)d_in[26], (const float*)d_in[27], MLP_, D_,
        g1_r, g1_i, nullptr, nullptr, nullptr, nullptr, nullptr, nullptr);

    // f2 + gate(g_m = chunk 5) + residual(x2) -> d_out (re, then im)
    float* out_re = (float*)d_out;
    float* out_im = out_re + MD;
    cgemm_kernel<4><<<gD, blk, 0, stream>>>(g1_r, g1_i, wf2_r, wf2_i,
        (const float*)d_in[30], (const float*)d_in[31], D_, MLP_,
        nullptr, nullptr, out_re, out_im,
        m_r + 5 * D_, m_i + 5 * D_, x2_r, x2_i);
}
